// NeuralPeronaMalik_14869176778973
// MI455X (gfx1250) — matrix operand order repaired and checked by reading, not compiled, not run on hardware
//
#include <hip/hip_runtime.h>
#include <math.h>

// ---------------------------------------------------------------------------
// NeuralPeronaMalik on MI455X (gfx1250): fused per-iteration kernel.
// 3x3 convs run as implicit GEMM on v_wmma_f32_16x16x32_bf16, decomposed as
// 9 shifted 1x1 convs; each K=32 fragment = two taps x 16 channels so the
// A-matrix is built with two ds_load_b128 per lane (channel-last LDS layout),
// zero im2col gather and zero integer divisions in the hot path.
// ---------------------------------------------------------------------------

typedef __bf16 bf16_t;
typedef __attribute__((ext_vector_type(16))) __bf16 v16bf;
typedef __attribute__((ext_vector_type(8)))  __bf16 v8bf;
typedef __attribute__((ext_vector_type(8)))  float  v8f;

constexpr int   Himg = 512, Wimg = 512, Bimg = 16;
constexpr int   TS = 16;           // output tile per workgroup
constexpr int   RX = TS + 8;       // x region, halo 4  (24)
constexpr int   RG = TS + 6;       // grads region, halo 3 (22)
constexpr int   R1 = TS + 4;       // h1 region, halo 2 (20)
constexpr int   R2 = TS + 2;       // h2 region, halo 1 (18)
constexpr float LAM = 0.1f;

struct SLds {
  alignas(32) bf16_t g[RG * RG][16];    // grads: ch 0..3 = N,S,E,W, 4..15 = 0
  alignas(32) bf16_t h1[R1 * R1][16];   // conv1 output (channel-last)
  alignas(32) bf16_t h2[R2 * R2][16];   // conv2 output (channel-last)
  alignas(16) float  x[RX][RX];         // fp32 x tile (edge-clamped)
  alignas(16) float  c[4][TS][TS];      // conv3 output (sigmoid), fp32
  alignas(16) bf16_t w1[16 * 4 * 9];    // bf16 weights, OIHW flattened
  alignas(16) bf16_t w2[16 * 16 * 9];
  alignas(16) bf16_t w3[4 * 16 * 9];
};

__device__ __forceinline__ v8f wmma_bf16(v16bf a, v16bf b, v8f c) {
  return __builtin_amdgcn_wmma_f32_16x16x32_bf16(false, a, false, b, (short)0,
                                                 c, false, false);
}

// A fragment for taps (2F, 2F+1): two 16-byte LDS loads per lane.
// pb = &in[(r*RIN + c0 + (lane&15))][ (lane<16)?0:8 ]  (channel-last layout).
template <int RIN, int F>
__device__ __forceinline__ v16bf load_A_taps(const bf16_t* __restrict__ pb) {
  union { v16bf v; v8bf h[2]; } a;
  constexpr int t0 = 2 * F;
  constexpr int o0 = ((t0 / 3) * RIN + (t0 % 3)) * 16;
  a.h[0] = *(const v8bf*)(pb + o0);
  if constexpr (2 * F + 1 < 9) {
    constexpr int t1 = 2 * F + 1;
    constexpr int o1 = ((t1 / 3) * RIN + (t1 % 3)) * 16;
    a.h[1] = *(const v8bf*)(pb + o1);
  } else {
    v8bf z = {};
    a.h[1] = z;            // keep WMMA inputs NaN-free for the padded tap
  }
  return a.v;
}

// B fragment for taps (2f, 2f+1): element i <-> tap 2f + (i >> 3), input
// channel (lane < 16 ? 0 : 8) + (i & 7), matching the A fragment layout.
__device__ __forceinline__ v16bf build_B_tap(const bf16_t* __restrict__ w,
                                             int Cin, int Nout, int f,
                                             int lane) {
  v16bf b;
  const int n = lane & 15;
  const int cb = (lane < 16) ? 0 : 8;
#pragma unroll
  for (int i = 0; i < 16; ++i) {
    bf16_t v = (bf16_t)0.0f;
    const int tap = 2 * f + (i >> 3);
    const int ch = cb + (i & 7);
    if (tap < 9 && ch < Cin && n < Nout) v = w[(n * Cin + ch) * 9 + tap];
    b[i] = v;
  }
  return b;
}

// Full 9-tap accumulation: 5 WMMAs, 9 ds_load_b128.
template <int RIN>
__device__ __forceinline__ v8f conv9(const bf16_t* __restrict__ pb,
                                     const v16bf Bf[5]) {
  v8f acc = {};
  acc = wmma_bf16(load_A_taps<RIN, 0>(pb), Bf[0], acc);
  acc = wmma_bf16(load_A_taps<RIN, 1>(pb), Bf[1], acc);
  acc = wmma_bf16(load_A_taps<RIN, 2>(pb), Bf[2], acc);
  acc = wmma_bf16(load_A_taps<RIN, 3>(pb), Bf[3], acc);
  acc = wmma_bf16(load_A_taps<RIN, 4>(pb), Bf[4], acc);
  return acc;
}

__global__ __launch_bounds__(256, 1) void pm_step_kernel(
    const float* __restrict__ xin, float* __restrict__ xout,
    const float* __restrict__ w1, const float* __restrict__ b1,
    const float* __restrict__ w2, const float* __restrict__ b2,
    const float* __restrict__ w3, const float* __restrict__ b3) {
  __shared__ SLds S;
  const int tid   = threadIdx.x;
  const int lane  = tid & 31;
  const int wid   = tid >> 5;
  const int m     = lane & 15;            // A-row / output-channel index
  const int half8 = (lane < 16) ? 0 : 8;  // channel-half of this lane
  const int gx0   = blockIdx.x * TS;
  const int gy0   = blockIdx.y * TS;
  const float* xb = xin + (size_t)blockIdx.z * Himg * Wimg;

  // ---- stage weights (fp32 -> bf16) into LDS -------------------------------
  for (int i = tid; i < 16 * 36; i += 256)  S.w1[i] = (bf16_t)w1[i];
  for (int i = tid; i < 16 * 144; i += 256) S.w2[i] = (bf16_t)w2[i];
  for (int i = tid; i < 4 * 144; i += 256)  S.w3[i] = (bf16_t)w3[i];

  // ---- load x tile with edge clamp (halo 4) --------------------------------
  for (int i = tid; i < RX * RX; i += 256) {
    const int r = i / RX, c = i % RX;
    const int gy = min(max(gy0 + r - 4, 0), Himg - 1);
    const int gx = min(max(gx0 + c - 4, 0), Wimg - 1);
    S.x[r][c] = xb[gy * Wimg + gx];
  }
  __syncthreads();

  // ---- gradients (edge-padded x), zeroed outside image, 16-ch padded -------
  for (int i = tid; i < RG * RG; i += 256) {
    const int r = i / RG, c = i % RG;              // x index = region idx + 1
    const int gy = gy0 + r - 3, gx = gx0 + c - 3;
    const float xc = S.x[r + 1][c + 1];
    const bool inb = (gy >= 0) & (gy < Himg) & (gx >= 0) & (gx < Wimg);
    v8bf lo = {};
    if (inb) {
      lo[0] = (bf16_t)(S.x[r][c + 1] - xc);        // N
      lo[1] = (bf16_t)(S.x[r + 2][c + 1] - xc);    // S
      lo[2] = (bf16_t)(S.x[r + 1][c + 2] - xc);    // E
      lo[3] = (bf16_t)(S.x[r + 1][c] - xc);        // W
    }
    v8bf hi = {};
    *(v8bf*)&S.g[i][0] = lo;
    *(v8bf*)&S.g[i][8] = hi;
  }
  __syncthreads();

  // ---- conv1: 4 -> 16 (ch-padded), ReLU, out region 20x20 ------------------
  {
    v16bf Bf[5];
#pragma unroll
    for (int f = 0; f < 5; ++f) Bf[f] = build_B_tap(S.w1, 4, 16, f, lane);
    const float bias = b1[m];
    for (int t = wid; t < R1 * 2; t += 8) {        // 20 rows x 2 chunks
      const int r  = t >> 1;
      const int c0 = (t & 1) ? (R1 - 16) : 0;      // overlapping chunks
      const bf16_t* pb = &S.g[r * RG + c0 + m][half8];
      v8f acc = conv9<RG>(pb, Bf);
      const int gy = gy0 + r - 2;
      const bool rowok = (gy >= 0) & (gy < Himg);
#pragma unroll
      for (int v = 0; v < 8; ++v) {
        const int mr = v + half8;
        const int cc = c0 + mr;
        const int gx = gx0 + cc - 2;
        const float hv = fmaxf(acc[v] + bias, 0.0f);
        const bool ok = rowok & (gx >= 0) & (gx < Wimg);
        S.h1[r * R1 + cc][m] = ok ? (bf16_t)hv : (bf16_t)0.0f;
      }
    }
  }
  __syncthreads();

  // ---- conv2: 16 -> 16, ReLU, out region 18x18 -----------------------------
  {
    v16bf Bf[5];
#pragma unroll
    for (int f = 0; f < 5; ++f) Bf[f] = build_B_tap(S.w2, 16, 16, f, lane);
    const float bias = b2[m];
    for (int t = wid; t < R2 * 2; t += 8) {        // 18 rows x 2 chunks
      const int r  = t >> 1;
      const int c0 = (t & 1) ? (R2 - 16) : 0;
      const bf16_t* pb = &S.h1[r * R1 + c0 + m][half8];
      v8f acc = conv9<R1>(pb, Bf);
      const int gy = gy0 + r - 1;
      const bool rowok = (gy >= 0) & (gy < Himg);
#pragma unroll
      for (int v = 0; v < 8; ++v) {
        const int mr = v + half8;
        const int cc = c0 + mr;
        const int gx = gx0 + cc - 1;
        const float hv = fmaxf(acc[v] + bias, 0.0f);
        const bool ok = rowok & (gx >= 0) & (gx < Wimg);
        S.h2[r * R2 + cc][m] = ok ? (bf16_t)hv : (bf16_t)0.0f;
      }
    }
  }
  __syncthreads();

  // ---- conv3: 16 -> 4, sigmoid, out 16x16 ----------------------------------
  {
    v16bf Bf[5];
#pragma unroll
    for (int f = 0; f < 5; ++f) Bf[f] = build_B_tap(S.w3, 16, 4, f, lane);
    const float bias = (m < 4) ? b3[m] : 0.0f;
    for (int t = wid; t < TS; t += 8) {            // 16 rows x 1 chunk
      const int r = t;
      const bf16_t* pb = &S.h2[r * R2 + m][half8];
      v8f acc = conv9<R2>(pb, Bf);
      if (m < 4) {
#pragma unroll
        for (int v = 0; v < 8; ++v) {
          const int mr = v + half8;
          const float z = acc[v] + bias;
          S.c[m][r][mr] = 1.0f / (1.0f + __expf(-z));
        }
      }
    }
  }
  __syncthreads();

  // ---- final update: x += lambda * sum(c_d * g_d), g in fp32 ---------------
  {
    const int ly = tid / TS, lx = tid % TS;
    const int gy = gy0 + ly, gx = gx0 + lx;
    const float xc = S.x[ly + 4][lx + 4];
    const float gN = S.x[ly + 3][lx + 4] - xc;
    const float gS = S.x[ly + 5][lx + 4] - xc;
    const float gE = S.x[ly + 4][lx + 5] - xc;
    const float gW = S.x[ly + 4][lx + 3] - xc;
    const float upd = S.c[0][ly][lx] * gN + S.c[1][ly][lx] * gS +
                      S.c[2][ly][lx] * gE + S.c[3][ly][lx] * gW;
    xout[((size_t)blockIdx.z * Himg + gy) * Wimg + gx] = xc + LAM * upd;
  }
}

// ---------------------------------------------------------------------------
// Host launcher: 10 iterations, ping-pong d_in -> ws -> d_out -> ws -> ...
// (iteration 9, the last, lands in d_out). Uses 16 MiB of d_ws.
// ---------------------------------------------------------------------------
extern "C" void kernel_launch(void* const* d_in, const int* in_sizes, int n_in,
                              void* d_out, int out_size, void* d_ws,
                              size_t ws_size, hipStream_t stream) {
  (void)in_sizes; (void)n_in; (void)out_size; (void)ws_size;
  const float* x0 = (const float*)d_in[0];
  const float* w1 = (const float*)d_in[1];
  const float* b1 = (const float*)d_in[2];
  const float* w2 = (const float*)d_in[3];
  const float* b2 = (const float*)d_in[4];
  const float* w3 = (const float*)d_in[5];
  const float* b3 = (const float*)d_in[6];
  float* out = (float*)d_out;
  float* ws  = (float*)d_ws;

  dim3 grid(Wimg / TS, Himg / TS, Bimg);
  dim3 blk(256);
  const float* src = x0;
  for (int it = 0; it < 10; ++it) {
    float* dst = (it & 1) ? out : ws;   // it=9 (last) -> d_out
    pm_step_kernel<<<grid, blk, 0, stream>>>(src, dst, w1, b1, w2, b2, w3, b3);
    src = dst;
  }
}
